// LSTMSwapModel_17540646437775
// MI455X (gfx1250) — matrix operand order repaired and checked by reading, not compiled, not run on hardware
//
#include <hip/hip_runtime.h>
#include <cstdint>
#include <cstddef>

#define B_   16
#define S_   100
#define V_   32000
#define D_   1024
#define H_   8
#define HID_ 4096
#define BS_  (B_*S_)

typedef __attribute__((ext_vector_type(16))) __bf16 v16bf;
typedef __attribute__((ext_vector_type(8)))  float  v8f;
typedef __attribute__((ext_vector_type(4)))  unsigned int u32x4;

typedef __attribute__((__vector_size__(16))) int v4i;        // builtin's pointee type
typedef __attribute__((address_space(1))) v4i *gv4p;         // global (AS1)
typedef __attribute__((address_space(3))) v4i *lv4p;         // LDS (AS3)

#if defined(__has_builtin)
# if __has_builtin(__builtin_amdgcn_global_load_async_to_lds_b128)
#  define USE_ASYNC 1
# endif
#endif
#ifndef USE_ASYNC
# define USE_ASYNC 0
#endif

union Frag { unsigned short u[16]; v16bf v; };

__device__ __forceinline__ unsigned short f2bf(float f){
  unsigned int x = __float_as_uint(f);
  unsigned int r = x + 0x7FFFu + ((x >> 16) & 1u);   // round-to-nearest-even
  return (unsigned short)(r >> 16);
}
__device__ __forceinline__ float sigf(float x){ return 1.f/(1.f+__expf(-x)); }

__device__ __forceinline__ void wait_async0(){
#if USE_ASYNC
# if __has_builtin(__builtin_amdgcn_s_wait_asynccnt)
  __builtin_amdgcn_s_wait_asynccnt(0);
# else
  asm volatile("s_wait_asynccnt 0x0" ::: "memory");
# endif
#endif
}

// ---------------------------------------------------------------------------
// Generic bf16 WMMA GEMM:  C[M][N] = A[M][K] * W[N][K]^T  (+bias, opt ReLU)
// 256 threads = 8 waves; 128x128 tile; K stepped by 64 (two wmma K-slabs,
// 16 v_wmma per barrier pair). Tiles staged via async global->LDS (ASYNCcnt).
// Wave grid 4(M) x 2(N); each wave owns 2x4 tiles of 16x16 -> 8 v8f accums.
// ---------------------------------------------------------------------------
__global__ __launch_bounds__(256) void k_gemm_bf16(
    const unsigned short* __restrict__ A,   // [M][K] bf16 (K mult of 64)
    const unsigned short* __restrict__ W,   // [N][K] bf16
    const float* __restrict__ bias,         // [N] or nullptr
    float* __restrict__ C,                  // [M][N]  (N multiple of 128)
    int M, int N, int K, int relu)
{
  __shared__ __align__(16) unsigned short As[128][72];  // 144B row stride (16B mult)
  __shared__ __align__(16) unsigned short Bs[128][72];
  const int tid  = threadIdx.x;
  const int lane = tid & 31;
  const int wave = tid >> 5;
  const int m0 = blockIdx.y * 128;
  const int n0 = blockIdx.x * 128;
  const int wm = (wave >> 1) * 32;          // wave M offset inside tile
  const int wn = (wave & 1) * 64;           // wave N offset inside tile
  const int hl = lane & 15;                 // half-lane index
  const int hi = lane >> 4;                 // which 16-lane half

  v8f zero = {0.f,0.f,0.f,0.f,0.f,0.f,0.f,0.f};
  v8f acc[2][4];
  for (int i = 0; i < 2; ++i) for (int j = 0; j < 4; ++j) acc[i][j] = zero;
  const u32x4 z4 = {0u,0u,0u,0u};

  for (int k0 = 0; k0 < K; k0 += 64) {
#if USE_ASYNC
    // 128 rows x 64 ushort = 16KB per tile; 1024 b128 transfers; 4 per thread
    for (int idx = tid; idx < 1024; idx += 256) {
      int r = idx >> 3, c = idx & 7;                    // c: 16B chunk in row
      int gr = m0 + r;
      if (gr < M)
        __builtin_amdgcn_global_load_async_to_lds_b128(
            (gv4p)(uintptr_t)(A + (size_t)gr * K + k0 + c * 8),
            (lv4p)(void*)&As[r][c * 8], 0, 0);
      else
        *(u32x4*)&As[r][c * 8] = z4;
      int gn = n0 + r;
      if (gn < N)
        __builtin_amdgcn_global_load_async_to_lds_b128(
            (gv4p)(uintptr_t)(W + (size_t)gn * K + k0 + c * 8),
            (lv4p)(void*)&Bs[r][c * 8], 0, 0);
      else
        *(u32x4*)&Bs[r][c * 8] = z4;
    }
    wait_async0();
#else
    const uint32_t* A32 = (const uint32_t*)A;
    const uint32_t* W32 = (const uint32_t*)W;
    for (int idx = tid; idx < 4096; idx += 256) {       // 128 rows x 32 dwords
      int r = idx >> 5, c2 = idx & 31;
      int gr = m0 + r;
      uint32_t val = 0u;
      if (gr < M) val = A32[(((size_t)gr * K + k0) >> 1) + c2];
      *(uint32_t*)&As[r][c2 * 2] = val;
      int gn = n0 + r;
      uint32_t wv = 0u;
      if (gn < N) wv = W32[(((size_t)gn * K + k0) >> 1) + c2];
      *(uint32_t*)&Bs[r][c2 * 2] = wv;
    }
#endif
    __syncthreads();

#pragma unroll
    for (int ks = 0; ks < 64; ks += 32) {
      // pack fragments per ISA 16-bit layouts
      Frag af[2], bf[4];
#pragma unroll
      for (int i = 0; i < 2; ++i) {
        int row = wm + i * 16 + hl;
        int kb  = ks + hi * 8;                 // lanes>=16 start at K=8
#pragma unroll
        for (int e = 0; e < 8; ++e) {
          af[i].u[e]     = As[row][kb + e];        // K = kb + e
          af[i].u[e + 8] = As[row][kb + 16 + e];   // K = kb + 16 + e
        }
      }
#pragma unroll
      for (int j = 0; j < 4; ++j) {
        int col = wn + j * 16 + hl;
        int kb  = ks + hi * 8;                 // lanes>=16 start at K=8
#pragma unroll
        for (int e = 0; e < 8; ++e) {
          bf[j].u[e]     = Bs[col][kb + e];
          bf[j].u[e + 8] = Bs[col][kb + 16 + e];
        }
      }
#pragma unroll
      for (int i = 0; i < 2; ++i)
#pragma unroll
        for (int j = 0; j < 4; ++j)
          acc[i][j] = __builtin_amdgcn_wmma_f32_16x16x32_bf16(
              false, af[i].v, false, bf[j].v, (short)0, acc[i][j], false, false);
    }
    __syncthreads();
  }

  // store: lane l, vgpr r -> M = r + 8*hi, N = hl
  for (int i = 0; i < 2; ++i)
    for (int j = 0; j < 4; ++j) {
      int col = n0 + wn + j * 16 + hl;
      float bv = bias ? bias[col] : 0.f;
#pragma unroll
      for (int r = 0; r < 8; ++r) {
        int row = m0 + wm + i * 16 + r + hi * 8;
        if (row < M) {
          float v = acc[i][j][r] + bv;
          if (relu) v = fmaxf(v, 0.f);
          C[(size_t)row * N + col] = v;
        }
      }
    }
}

// ---------------------------------------------------------------------------
// weight prep
// ---------------------------------------------------------------------------
// direct: dst[n][k] = bf16(src[n][k])  (src already [N][K], coalesced)
__global__ void k_prep_w(const float* __restrict__ src, unsigned short* __restrict__ dst,
                         int K, int N, int srcKN){
  size_t i = (size_t)blockIdx.x * 256 + threadIdx.x;
  if (i >= (size_t)N * K) return;
  int n = (int)(i / K), k = (int)(i % K);
  float v = srcKN ? src[(size_t)k * N + n] : src[i];
  dst[i] = f2bf(v);
}

// transposing: dst[n][k] = bf16(src[k][n]) via 32x32 LDS tile (both sides coalesced)
__global__ __launch_bounds__(256) void k_prep_wT(const float* __restrict__ src,
                                                 unsigned short* __restrict__ dst,
                                                 int K, int N){
  __shared__ float t[32][33];
  int n0 = blockIdx.x * 32, k0 = blockIdx.y * 32;
  int tx = threadIdx.x & 31, ty = threadIdx.x >> 5;    // 8-row passes
  for (int r = ty; r < 32; r += 8)
    t[r][tx] = src[(size_t)(k0 + r) * N + n0 + tx];
  __syncthreads();
  for (int r = ty; r < 32; r += 8)
    dst[(size_t)(n0 + r) * K + k0 + tx] = f2bf(t[tx][r]);
}

__global__ void k_cvt(const float* __restrict__ s, unsigned short* __restrict__ d, size_t n){
  size_t i = (size_t)blockIdx.x * 256 + threadIdx.x;
  if (i < n) d[i] = f2bf(s[i]);
}

// embedding gather * sqrt(D)
__global__ __launch_bounds__(256) void k_embed(const int* __restrict__ ids,
                                               const float* __restrict__ emb,
                                               float* __restrict__ xs,
                                               unsigned short* __restrict__ xsbf){
  int row = blockIdx.x;
  int id  = ids[row];
#pragma unroll
  for (int e = 0; e < 4; ++e) {
    int d = threadIdx.x + e * 256;
    float v = emb[(size_t)id * D_ + d] * 32.0f;   // sqrt(1024)
    xs[(size_t)row * D_ + d] = v;
    xsbf[(size_t)row * D_ + d] = f2bf(v);
  }
}

// attention with ALiBi: one block per (b,h,s); dh = 128
__global__ __launch_bounds__(128) void k_attn(const float* __restrict__ q,
                                              const float* __restrict__ k,
                                              const float* __restrict__ v,
                                              float* __restrict__ o){
  int idx = blockIdx.x;
  int s = idx % S_; int h = (idx / S_) % H_; int b = idx / (S_ * H_);
  __shared__ float qs[128], sc[128], red[128];
  int tid = threadIdx.x;
  size_t qoff = ((size_t)(b * S_ + s) * D_) + h * 128;
  qs[tid] = q[qoff + tid];
  __syncthreads();
  float scv = -1e30f;
  if (tid < S_) {
    size_t koff = ((size_t)(b * S_ + tid) * D_) + h * 128;
    float d = 0.f;
    for (int e = 0; e < 128; ++e) d += qs[e] * k[koff + e];
    scv = d * 0.08838834764831845f + exp2f(-(float)(h + 1)) * (float)(tid - s);
  }
  red[tid] = scv; __syncthreads();
  for (int off = 64; off; off >>= 1) { if (tid < off) red[tid] = fmaxf(red[tid], red[tid + off]); __syncthreads(); }
  float mx = red[0]; __syncthreads();
  float e = (tid < S_) ? __expf(scv - mx) : 0.f;
  sc[tid] = e; red[tid] = e; __syncthreads();
  for (int off = 64; off; off >>= 1) { if (tid < off) red[tid] += red[tid + off]; __syncthreads(); }
  float inv = 1.f / red[0];
  float acc = 0.f;
  for (int t = 0; t < S_; ++t)
    acc += sc[t] * inv * v[((size_t)(b * S_ + t) * D_) + h * 128 + tid];
  o[qoff + tid] = acc;
}

// y = LN(x (+ r)) * s + b ; optional bf16 copy ; in-place safe per-row
__global__ __launch_bounds__(256) void k_add_ln(const float* __restrict__ x,
                                                const float* __restrict__ r,
                                                const float* __restrict__ sc,
                                                const float* __restrict__ bi,
                                                float* __restrict__ y,
                                                unsigned short* __restrict__ ybf){
  int row = blockIdx.x, tid = threadIdx.x;
  __shared__ float r1[256], r2[256];
  float v[4]; float s1 = 0.f, s2 = 0.f;
#pragma unroll
  for (int e = 0; e < 4; ++e) {
    size_t ix = (size_t)row * D_ + tid + e * 256;
    float a = x[ix];
    if (r) a += r[ix];
    v[e] = a; s1 += a; s2 += a * a;
  }
  r1[tid] = s1; r2[tid] = s2; __syncthreads();
  for (int off = 128; off; off >>= 1) { if (tid < off) { r1[tid] += r1[tid + off]; r2[tid] += r2[tid + off]; } __syncthreads(); }
  float mu = r1[0] * (1.f / D_);
  float var = r2[0] * (1.f / D_) - mu * mu;
  float rs = rsqrtf(var + 1e-5f);
#pragma unroll
  for (int e = 0; e < 4; ++e) {
    int d = tid + e * 256;
    float ov = (v[e] - mu) * rs * sc[d] + bi[d];
    y[(size_t)row * D_ + d] = ov;
    if (ybf) ybf[(size_t)row * D_ + d] = f2bf(ov);
  }
}

__global__ void k_lstm_init(float* c, unsigned short* hbf){
  int i = blockIdx.x * 256 + threadIdx.x;
  if (i < B_ * D_) { c[i] = 0.f; hbf[i] = 0; }
}

// gates i,f,g,o at offsets 0,D,2D,3D
__global__ void k_lstm_point(const float* __restrict__ gin, const float* __restrict__ grec,
                             const float* __restrict__ bih, const float* __restrict__ bhh,
                             float* __restrict__ c, float* __restrict__ hseq,
                             unsigned short* __restrict__ hbf, int t){
  int i = blockIdx.x * 256 + threadIdx.x;
  if (i >= B_ * D_) return;
  int b = i / D_, d = i % D_;
  size_t gx = ((size_t)b * S_ + t) * (4 * D_);
  size_t gr = (size_t)b * (4 * D_);
  float gi = gin[gx + d]            + grec[gr + d]            + bih[d]            + bhh[d];
  float gf = gin[gx + D_ + d]       + grec[gr + D_ + d]       + bih[D_ + d]       + bhh[D_ + d];
  float gg = gin[gx + 2 * D_ + d]   + grec[gr + 2 * D_ + d]   + bih[2 * D_ + d]   + bhh[2 * D_ + d];
  float go = gin[gx + 3 * D_ + d]   + grec[gr + 3 * D_ + d]   + bih[3 * D_ + d]   + bhh[3 * D_ + d];
  float cn = sigf(gf) * c[i] + sigf(gi) * tanhf(gg);
  float h  = sigf(go) * tanhf(cn);
  c[i] = cn;
  hseq[((size_t)b * S_ + t) * D_ + d] = h;
  hbf[i] = f2bf(h);
}

// out[r] = dot(x[r,:], w) + b0
__global__ __launch_bounds__(256) void k_rowdot(const float* __restrict__ x,
                                                const float* __restrict__ w,
                                                const float* __restrict__ b,
                                                float* __restrict__ out){
  int r = blockIdx.x, tid = threadIdx.x;
  __shared__ float red[256];
  float a = 0.f;
  for (int d = tid; d < D_; d += 256) a += x[(size_t)r * D_ + d] * w[d];
  red[tid] = a; __syncthreads();
  for (int off = 128; off; off >>= 1) { if (tid < off) red[tid] += red[tid + off]; __syncthreads(); }
  if (tid == 0) out[r] = red[0] + b[0];
}

__device__ __forceinline__ float urand(unsigned int x){
  x ^= x >> 16; x *= 0x7feb352du; x ^= x >> 15; x *= 0x846ca68bu; x ^= x >> 16;
  return (float)(x >> 8) * (1.0f / 16777216.f) + 1e-9f;
}

// hard gumbel over axis S -> one-hot at argmax of perturbed logits
__global__ __launch_bounds__(128) void k_gumbel(const float* __restrict__ logits,
                                                float* __restrict__ swp, unsigned int seed){
  int b = blockIdx.x, tid = threadIdx.x;
  __shared__ float rv[128]; __shared__ int ri[128];
  float z = -1e30f;
  if (tid < S_) {
    float u = urand(seed ^ (unsigned)(b * 1315423911u) ^ (unsigned)(tid * 2654435761u));
    z = logits[b * S_ + tid] - logf(-logf(u));
  }
  rv[tid] = z; ri[tid] = tid; __syncthreads();
  for (int off = 64; off; off >>= 1) {
    if (tid < off && rv[tid + off] > rv[tid]) { rv[tid] = rv[tid + off]; ri[tid] = ri[tid + off]; }
    __syncthreads();
  }
  if (tid < S_) swp[b * S_ + tid] = (tid == ri[0]) ? 1.f : 0.f;
}

__global__ void k_cumsum(const float* __restrict__ fromp, const float* __restrict__ top,
                         float* fcs1, float* fcs2, float* tcs1, float* tcs2){
  int b = threadIdx.x;
  if (b >= B_) return;
  float a = 0.f, c = 0.f;
  for (int s = 0; s < S_; ++s) {
    a += top[b * S_ + s];  c += fromp[b * S_ + s];
    fcs2[b * S_ + s] = a;  fcs1[b * S_ + s] = 1.f - a;
    tcs2[b * S_ + s] = c;  tcs1[b * S_ + s] = 1.f - c;
  }
}

// ys[b,s,:] = sum_t M[b,s,t] * xs[b,t,:]  (lb+mb+val+rb construction)
__global__ __launch_bounds__(256) void k_permute(const float* __restrict__ xs,
                                                 const float* __restrict__ fromp,
                                                 const float* __restrict__ top,
                                                 const float* __restrict__ fcs1,
                                                 const float* __restrict__ fcs2,
                                                 const float* __restrict__ tcs1,
                                                 const float* __restrict__ tcs2,
                                                 float* __restrict__ ys,
                                                 unsigned short* __restrict__ ysbf){
  int rs = blockIdx.x; int s = rs % S_; int b = rs / S_;
  int tid = threadIdx.x;
  __shared__ float coef[128];
  if (tid < S_) {
    int t = tid;
    float m = top[b * S_ + s] * fromp[b * S_ + t];                       // val
    if (t == s) {
      m += fcs1[b * S_ + s] * tcs1[b * S_ + s];                          // lb
      if (s > 0) m += tcs2[b * S_ + s - 1] * fcs2[b * S_ + s - 1];       // rb
    }
    if (t == s - 1) m += fcs2[b * S_ + t] * tcs1[b * S_ + t];            // mb (shift_dn)
    if (t == s + 1) m += fcs1[b * S_ + s] * tcs2[b * S_ + s];            // mb (shift_up)
    coef[t] = m;
  }
  __syncthreads();
  float acc[4] = {0.f, 0.f, 0.f, 0.f};
  for (int t = 0; t < S_; ++t) {
    float m = coef[t];
    const float* xr = &xs[((size_t)(b * S_ + t)) * D_];
#pragma unroll
    for (int e = 0; e < 4; ++e) acc[e] += m * xr[tid + e * 256];
  }
  size_t ro = ((size_t)(b * S_ + s)) * D_;
#pragma unroll
  for (int e = 0; e < 4; ++e) {
    ys[ro + tid + e * 256] = acc[e];
    ysbf[ro + tid + e * 256] = f2bf(acc[e]);
  }
}

// ---------------------------------------------------------------------------
// host-side orchestration helpers
// ---------------------------------------------------------------------------
static void prep(hipStream_t st, const float* src, unsigned short* dst, int K, int N, int srcKN){
  if (srcKN) {
    dim3 g(N / 32, K / 32);
    k_prep_wT<<<g, 256, 0, st>>>(src, dst, K, N);
  } else {
    size_t n = (size_t)K * N;
    k_prep_w<<<(unsigned)((n + 255) / 256), 256, 0, st>>>(src, dst, K, N, 0);
  }
}
static void gemm(hipStream_t st, const unsigned short* A, const unsigned short* W,
                 const float* bias, float* C, int M, int N, int K, int relu){
  dim3 g(N / 128, (M + 127) / 128);
  k_gemm_bf16<<<g, 256, 0, st>>>(A, W, bias, C, M, N, K, relu);
}
static void cvt(hipStream_t st, const float* s, unsigned short* d, size_t n){
  k_cvt<<<(unsigned)((n + 255) / 256), 256, 0, st>>>(s, d, n);
}

struct Tmps {
  unsigned short* wbf; float* q; float* kk; float* vv; float* tO; float* tP;
  unsigned short* bfT1; unsigned short* bfT2; float* gin; unsigned short* ginbf;
};

static void run_decoder(hipStream_t st, void* const* din, int pb,
                        const float* xin, const unsigned short* xinbf,
                        float* xout, unsigned short* xoutbf, const Tmps& T){
  const float* wq  = (const float*)din[pb + 0];  const float* wqb = (const float*)din[pb + 1];
  const float* wk  = (const float*)din[pb + 2];  const float* wkb = (const float*)din[pb + 3];
  const float* wv  = (const float*)din[pb + 4];  const float* wvb = (const float*)din[pb + 5];
  const float* wo  = (const float*)din[pb + 6];  const float* wob = (const float*)din[pb + 7];
  const float* w1  = (const float*)din[pb + 8];  const float* b1  = (const float*)din[pb + 9];
  const float* w2  = (const float*)din[pb + 10]; const float* b2  = (const float*)din[pb + 11];
  const float* l1s = (const float*)din[pb + 12]; const float* l1b = (const float*)din[pb + 13];
  const float* l2s = (const float*)din[pb + 14]; const float* l2b = (const float*)din[pb + 15];

  prep(st, wq, T.wbf, D_, D_, 1); gemm(st, xinbf, T.wbf, wqb, T.q,  BS_, D_, D_, 0);
  prep(st, wk, T.wbf, D_, D_, 1); gemm(st, xinbf, T.wbf, wkb, T.kk, BS_, D_, D_, 0);
  prep(st, wv, T.wbf, D_, D_, 1); gemm(st, xinbf, T.wbf, wvb, T.vv, BS_, D_, D_, 0);
  k_attn<<<B_ * H_ * S_, 128, 0, st>>>(T.q, T.kk, T.vv, T.tO);
  cvt(st, T.tO, T.bfT1, (size_t)BS_ * D_);
  prep(st, wo, T.wbf, D_, D_, 1); gemm(st, T.bfT1, T.wbf, wob, T.tP, BS_, D_, D_, 0);
  k_add_ln<<<BS_, 256, 0, st>>>(xin, T.tP, l1s, l1b, T.tO, T.bfT2);     // x1
  prep(st, w1, T.wbf, D_, HID_, 1); gemm(st, T.bfT2, T.wbf, b1, T.gin, BS_, HID_, D_, 1);
  cvt(st, T.gin, T.ginbf, (size_t)BS_ * HID_);
  prep(st, w2, T.wbf, HID_, D_, 1); gemm(st, T.ginbf, T.wbf, b2, T.tP, BS_, D_, HID_, 0);
  k_add_ln<<<BS_, 256, 0, st>>>(T.tO, T.tP, l2s, l2b, xout, xoutbf);
}

static void run_lstm_stack(hipStream_t st, void* const* din, int base,
                           const float* xin, const unsigned short* xinbf,
                           float* ppA, unsigned short* ppAbf,
                           float* ppB, unsigned short* ppBbf,
                           unsigned short* wbf, unsigned short* whhbf,
                           float* gin, float* grec, float* cst, unsigned short* hbf){
  const float* cur = xin; const unsigned short* curbf = xinbf;
  for (int L = 0; L < 2; ++L) {
    int pb = base + 6 * L;
    const float* wih = (const float*)din[pb + 0]; const float* whh = (const float*)din[pb + 1];
    const float* bih = (const float*)din[pb + 2]; const float* bhh = (const float*)din[pb + 3];
    const float* lns = (const float*)din[pb + 4]; const float* lnb = (const float*)din[pb + 5];
    float* hs = (L == 0) ? ppA : ppB;
    unsigned short* hsbf = (L == 0) ? ppAbf : ppBbf;
    prep(st, wih, wbf, D_, 4 * D_, 0);                 // wih is [4D][D] = [N][K]
    gemm(st, curbf, wbf, nullptr, gin, BS_, 4 * D_, D_, 0);
    prep(st, whh, whhbf, D_, 4 * D_, 0);
    k_lstm_init<<<(B_ * D_ + 255) / 256, 256, 0, st>>>(cst, hbf);
    for (int t = 0; t < S_; ++t) {
      gemm(st, hbf, whhbf, nullptr, grec, B_, 4 * D_, D_, 0);
      k_lstm_point<<<(B_ * D_ + 255) / 256, 256, 0, st>>>(gin, grec, bih, bhh, cst, hs, hbf, t);
    }
    k_add_ln<<<BS_, 256, 0, st>>>(hs, nullptr, lns, lnb, hs, hsbf);   // in-place LN
    cur = hs; curbf = hsbf;
  }
  (void)cur; (void)curbf;   // result lives in ppB/ppBbf
}

// input indices (setup_inputs flatten order)
#define IN_IDS 0
#define IN_EMB 1
#define IN_TI1 2
#define IN_TO1 18
#define IN_TO2 34
#define IN_L1  50
#define IN_L2  62
#define IN_O1W 74
#define IN_O1B 75
#define IN_O2W 76
#define IN_O2B 77
#define IN_OUTW 78
#define IN_OUTB 79

extern "C" void kernel_launch(void* const* d_in, const int* in_sizes, int n_in,
                              void* d_out, int out_size, void* d_ws, size_t ws_size,
                              hipStream_t stream){
  (void)in_sizes; (void)n_in; (void)out_size; (void)ws_size;
  const int*   ids = (const int*)d_in[IN_IDS];
  const float* emb = (const float*)d_in[IN_EMB];

  char* base = (char*)d_ws;
  size_t off = 0;
  auto alloc = [&](size_t bytes) -> void* {
    void* p = base + off;
    off += (bytes + 255) & ~(size_t)255;
    return p;
  };
  const size_t BSD = (size_t)BS_ * D_;

  float* xs            = (float*)alloc(sizeof(float) * BSD);
  unsigned short* xsbf = (unsigned short*)alloc(2 * BSD);
  float* d1            = (float*)alloc(sizeof(float) * BSD);
  unsigned short* d1bf = (unsigned short*)alloc(2 * BSD);
  float* ppA           = (float*)alloc(sizeof(float) * BSD);
  unsigned short* ppAbf= (unsigned short*)alloc(2 * BSD);
  float* ppB           = (float*)alloc(sizeof(float) * BSD);
  unsigned short* ppBbf= (unsigned short*)alloc(2 * BSD);
  float* r1            = (float*)alloc(sizeof(float) * BSD);
  unsigned short* r1bf = (unsigned short*)alloc(2 * BSD);
  float* ys            = (float*)alloc(sizeof(float) * BSD);
  unsigned short* ysbf = (unsigned short*)alloc(2 * BSD);

  Tmps T;
  T.q    = (float*)alloc(sizeof(float) * BSD);
  T.kk   = (float*)alloc(sizeof(float) * BSD);
  T.vv   = (float*)alloc(sizeof(float) * BSD);
  T.tO   = (float*)alloc(sizeof(float) * BSD);
  T.tP   = (float*)alloc(sizeof(float) * BSD);
  T.bfT1 = (unsigned short*)alloc(2 * BSD);
  T.bfT2 = (unsigned short*)alloc(2 * BSD);
  T.gin  = (float*)alloc(sizeof(float) * (size_t)BS_ * HID_);
  T.ginbf= (unsigned short*)alloc(2 * (size_t)BS_ * HID_);
  T.wbf  = (unsigned short*)alloc(2 * (size_t)D_ * V_);          // largest weight (out_w)
  unsigned short* whhbf = (unsigned short*)alloc(2 * (size_t)4 * D_ * D_);

  float* grec = (float*)alloc(sizeof(float) * B_ * 4 * D_);
  float* cst  = (float*)alloc(sizeof(float) * B_ * D_);
  unsigned short* hbf = (unsigned short*)alloc(2 * (size_t)B_ * D_);
  float* n1o  = (float*)alloc(sizeof(float) * BS_);
  float* n2o  = (float*)alloc(sizeof(float) * BS_);
  float* sw1  = (float*)alloc(sizeof(float) * BS_);
  float* sw2  = (float*)alloc(sizeof(float) * BS_);
  float* fcs1 = (float*)alloc(sizeof(float) * BS_);
  float* fcs2 = (float*)alloc(sizeof(float) * BS_);
  float* tcs1 = (float*)alloc(sizeof(float) * BS_);
  float* tcs2 = (float*)alloc(sizeof(float) * BS_);

  // 1. embedding
  k_embed<<<BS_, 256, 0, stream>>>(ids, emb, xs, xsbf);

  // 2. shared first decoder (ti1) -- reference uses ti1 for BOTH branches
  run_decoder(stream, d_in, IN_TI1, xs, xsbf, d1, d1bf, T);

  // 3. branch 1: lstm1 stack -> to1 decoder -> o1 head
  run_lstm_stack(stream, d_in, IN_L1, d1, d1bf, ppA, ppAbf, ppB, ppBbf,
                 T.wbf, whhbf, T.gin, grec, cst, hbf);
  run_decoder(stream, d_in, IN_TO1, ppB, ppBbf, r1, r1bf, T);
  k_rowdot<<<BS_, 256, 0, stream>>>(r1, (const float*)d_in[IN_O1W], (const float*)d_in[IN_O1B], n1o);

  // 4. branch 2: lstm2 stack -> to2 decoder -> o2 head
  run_lstm_stack(stream, d_in, IN_L2, d1, d1bf, ppA, ppAbf, ppB, ppBbf,
                 T.wbf, whhbf, T.gin, grec, cst, hbf);
  run_decoder(stream, d_in, IN_TO2, ppB, ppBbf, r1, r1bf, T);
  k_rowdot<<<BS_, 256, 0, stream>>>(r1, (const float*)d_in[IN_O2W], (const float*)d_in[IN_O2B], n2o);

  // 5. gumbel-hard swaps, cumsums, permutation
  k_gumbel<<<B_, 128, 0, stream>>>(n1o, sw1, 0x12345u);
  k_gumbel<<<B_, 128, 0, stream>>>(n2o, sw2, 0x6789Bu);
  k_cumsum<<<1, 32, 0, stream>>>(sw1, sw2, fcs1, fcs2, tcs1, tcs2);
  k_permute<<<BS_, 256, 0, stream>>>(xs, sw1, sw2, fcs1, fcs2, tcs1, tcs2, ys, ysbf);

  // 6. vocab projection (dominant GEMM: 1600 x 32000 x 1024)
  prep(stream, (const float*)d_in[IN_OUTW], T.wbf, D_, V_, 1);
  gemm(stream, ysbf, T.wbf, (const float*)d_in[IN_OUTB], (float*)d_out, BS_, V_, D_, 0);
}
